// MoE_MLP_55087250539083
// MI455X (gfx1250) — matrix operand order repaired and checked by reading, not compiled, not run on hardware
//
#include <hip/hip_runtime.h>
#include <hip/hip_bf16.h>
#include <stdint.h>

// Problem constants (match reference)
#define S_TOK   2048
#define DMODEL  1024
#define DEXPERT 2816
#define NEXP    8
#define KTOP    2

typedef __attribute__((ext_vector_type(16))) __bf16 v16bf;
typedef __attribute__((ext_vector_type(8)))  float  v8f;

union FragB { v16bf v; uint4 q[2]; };

// ---------------------------------------------------------------------------
// Kernel 0: zero per-expert counters
// ---------------------------------------------------------------------------
__global__ void k_init(int* __restrict__ cnt) {
    if (threadIdx.x < NEXP) cnt[threadIdx.x] = 0;
}

// ---------------------------------------------------------------------------
// Kernel 1: fp32 -> bf16 weight conversion (weights then live in 192MB L2)
// ---------------------------------------------------------------------------
__global__ void k_cvt(const float* __restrict__ src, __bf16* __restrict__ dst, int n4) {
    int i = blockIdx.x * blockDim.x + threadIdx.x;
    int stride = gridDim.x * blockDim.x;
    for (; i < n4; i += stride) {
        float4 v = ((const float4*)src)[i];
        union { __bf16 h[4]; uint2 u; } pk;
        pk.h[0] = (__bf16)v.x; pk.h[1] = (__bf16)v.y;
        pk.h[2] = (__bf16)v.z; pk.h[3] = (__bf16)v.w;
        ((uint2*)dst)[i] = pk.u;
    }
}

// ---------------------------------------------------------------------------
// Kernel 2: RMSNorm + router (top-2 softmax) + per-expert token bucketing
// One block (128 threads) per token.
// ---------------------------------------------------------------------------
__global__ __launch_bounds__(128) void k_router(
    const float* __restrict__ x, const float* __restrict__ gate_w,
    const float* __restrict__ rms_w, __bf16* __restrict__ wsY,
    int* __restrict__ cnt, int* __restrict__ tokbuf, float* __restrict__ wbuf)
{
    const int t   = blockIdx.x;
    const int tid = threadIdx.x;
    __shared__ float sred[128];
    __shared__ float slog[NEXP][128];
    __shared__ float sscale;

    const float4* x4 = (const float4*)(x + (size_t)t * DMODEL);
    float4 xv0 = x4[tid];
    float4 xv1 = x4[tid + 128];
    float ss = xv0.x*xv0.x + xv0.y*xv0.y + xv0.z*xv0.z + xv0.w*xv0.w
             + xv1.x*xv1.x + xv1.y*xv1.y + xv1.z*xv1.z + xv1.w*xv1.w;
    sred[tid] = ss;
    __syncthreads();
    for (int off = 64; off > 0; off >>= 1) {
        if (tid < off) sred[tid] += sred[tid + off];
        __syncthreads();
    }
    if (tid == 0) sscale = rsqrtf(sred[0] * (1.0f / DMODEL) + 1e-6f);
    __syncthreads();
    const float sc = sscale;

    const float4* rw4 = (const float4*)rms_w;
    float4 r0 = rw4[tid], r1 = rw4[tid + 128];
    float4 y0, y1;
    y0.x = xv0.x * sc * r0.x; y0.y = xv0.y * sc * r0.y;
    y0.z = xv0.z * sc * r0.z; y0.w = xv0.w * sc * r0.w;
    y1.x = xv1.x * sc * r1.x; y1.y = xv1.y * sc * r1.y;
    y1.z = xv1.z * sc * r1.z; y1.w = xv1.w * sc * r1.w;

    // store normalized activations as bf16 (A-matrix source for WMMA)
    __bf16* yrow = wsY + (size_t)t * DMODEL;
    union { __bf16 h[4]; uint2 u; } pk;
    pk.h[0] = (__bf16)y0.x; pk.h[1] = (__bf16)y0.y;
    pk.h[2] = (__bf16)y0.z; pk.h[3] = (__bf16)y0.w;
    ((uint2*)yrow)[tid] = pk.u;
    pk.h[0] = (__bf16)y1.x; pk.h[1] = (__bf16)y1.y;
    pk.h[2] = (__bf16)y1.z; pk.h[3] = (__bf16)y1.w;
    ((uint2*)yrow)[tid + 128] = pk.u;

    // router logits (f32, as in reference)
    #pragma unroll
    for (int e = 0; e < NEXP; ++e) {
        const float4* g4 = (const float4*)(gate_w + (size_t)e * DMODEL);
        float4 g0 = g4[tid], g1 = g4[tid + 128];
        float p = y0.x*g0.x + y0.y*g0.y + y0.z*g0.z + y0.w*g0.w
                + y1.x*g1.x + y1.y*g1.y + y1.z*g1.z + y1.w*g1.w;
        slog[e][tid] = p;
    }
    __syncthreads();
    if (tid < NEXP) {
        float s = 0.f;
        for (int i = 0; i < 128; ++i) s += slog[tid][i];
        slog[tid][0] = s;
    }
    __syncthreads();
    if (tid == 0) {
        float v[NEXP];
        #pragma unroll
        for (int e = 0; e < NEXP; ++e) v[e] = slog[e][0];
        int e0 = 0;
        for (int e = 1; e < NEXP; ++e) if (v[e] > v[e0]) e0 = e;
        int e1 = -1;
        for (int e = 0; e < NEXP; ++e)
            if (e != e0 && (e1 < 0 || v[e] > v[e1])) e1 = e;
        float ex  = __expf(v[e1] - v[e0]);
        float den = 1.0f + ex;
        float w0 = 1.0f / den, w1 = ex / den;
        int p0 = atomicAdd(&cnt[e0], 1);
        tokbuf[(size_t)e0 * S_TOK + p0] = (t << 1);
        wbuf  [(size_t)e0 * S_TOK + p0] = w0;
        int p1 = atomicAdd(&cnt[e1], 1);
        tokbuf[(size_t)e1 * S_TOK + p1] = (t << 1) | 1;
        wbuf  [(size_t)e1 * S_TOK + p1] = w1;
    }
}

// ---------------------------------------------------------------------------
// Kernel 3: fused per-expert SwiGLU with bf16 WMMA.
// grid = (S/16 tiles, E experts), 256 threads = 8 waves.
// DE processed in 2 chunks of 1408 so the h staging tile stays ~45KB of LDS.
// ---------------------------------------------------------------------------
__global__ __launch_bounds__(256) void k_expert(
    const __bf16* __restrict__ wsY,  const __bf16* __restrict__ wsW1,
    const __bf16* __restrict__ wsW2, const __bf16* __restrict__ wsW3,
    const int* __restrict__ cnt, const int* __restrict__ tokbuf,
    const float* __restrict__ wbuf, float* __restrict__ contrib)
{
    const int e        = blockIdx.y;
    const int nTok     = cnt[e];
    const int tileBase = blockIdx.x * 16;
    if (tileBase >= nTok) return;              // block-uniform exit

    const int tid  = threadIdx.x;
    const int wave = tid >> 5;
    const int lane = tid & 31;
    const int half = lane >> 4;                // 0: lanes 0-15, 1: lanes 16-31
    const int l16  = lane & 15;

    __shared__ __align__(16) __bf16 lH[16][1416]; // 16 x (1408+8 pad) bf16
    __shared__ int   lTok[16];
    __shared__ float lWgt[16];
    __shared__ int   lValid[16];

    if (tid < 16) {
        int g   = tileBase + tid;
        int val = (g < nTok);
        lTok[tid]   = tokbuf[(size_t)e * S_TOK + (val ? g : tileBase)];
        lWgt[tid]   = val ? wbuf[(size_t)e * S_TOK + g] : 0.0f;
        lValid[tid] = val;
    }
    __syncthreads();

    // Per-lane A-row base: A fragment row M == lane&15 (gfx1250 16-bit A layout)
    const __bf16* yRow = wsY + (size_t)(lTok[l16] >> 1) * DMODEL;
    const __bf16* w1e  = wsW1 + (size_t)e * DEXPERT * DMODEL;
    const __bf16* w2e  = wsW2 + (size_t)e * DEXPERT * DMODEL;
    const __bf16* w3e  = wsW3 + (size_t)e * DMODEL  * DEXPERT;

    const v8f vzero = {};
    v8f accO[8];
    #pragma unroll
    for (int i = 0; i < 8; ++i) accO[i] = vzero;

    for (int chunk = 0; chunk < 2; ++chunk) {
        const int nbase = chunk * 1408;

        // -------- phase A: up/gate GEMM + SwiGLU into lH --------
        for (int ntile = 0; ntile < 11; ++ntile) {
            const int n0 = nbase + wave * 176 + ntile * 16;
            v8f accU = vzero, accG = vzero;
            for (int k0 = 0; k0 < DMODEL; k0 += 32) {
                FragB a, b1, b2;
                // A: 16 contiguous K at k0+16*half (32B)
                const uint4* pa = (const uint4*)(yRow + k0 + 16 * half);
                a.q[0] = pa[0];
                a.q[1] = pa[1];
                // B: lane = column n, 16 contiguous K at k0+16*half (32B)
                const uint4* pb1 = (const uint4*)(w1e + (size_t)(n0 + l16) * DMODEL + k0 + 16 * half);
                b1.q[0] = pb1[0]; b1.q[1] = pb1[1];
                const uint4* pb2 = (const uint4*)(w2e + (size_t)(n0 + l16) * DMODEL + k0 + 16 * half);
                b2.q[0] = pb2[0]; b2.q[1] = pb2[1];
                accU = __builtin_amdgcn_wmma_f32_16x16x32_bf16(false, a.v, false, b1.v,
                                                               (short)0, accU, false, false);
                accG = __builtin_amdgcn_wmma_f32_16x16x32_bf16(false, a.v, false, b2.v,
                                                               (short)0, accG, false, false);
            }
            const int nl = n0 - nbase;
            #pragma unroll
            for (int i = 0; i < 8; ++i) {
                float up = accU[i], gt = accG[i];
                float hv = (up / (1.0f + __expf(-up))) * gt;   // silu(up) * gate
                lH[i + 8 * half][nl + l16] = (__bf16)hv;       // C row M = i + 8*half, col N = l16
            }
        }
        __syncthreads();

        // -------- phase B: out += h_chunk @ W3^T_chunk --------
        const uint4* rowH = (const uint4*)&lH[l16][0];
        #pragma unroll
        for (int mt = 0; mt < 8; ++mt) {
            const int m0 = wave * 128 + mt * 16;
            v8f acc = accO[mt];
            for (int kl = 0; kl < 1408; kl += 32) {
                FragB a, b;
                const int eo = (kl + 16 * half) >> 3;   // 16B units
                a.q[0] = rowH[eo];
                a.q[1] = rowH[eo + 1];
                const uint4* pb = (const uint4*)(w3e + (size_t)(m0 + l16) * DEXPERT
                                                 + nbase + kl + 16 * half);
                b.q[0] = pb[0]; b.q[1] = pb[1];
                acc = __builtin_amdgcn_wmma_f32_16x16x32_bf16(false, a.v, false, b.v,
                                                              (short)0, acc, false, false);
            }
            accO[mt] = acc;
        }
        __syncthreads();   // lH reused next chunk
    }

    // -------- scatter weighted contributions (deterministic slot) --------
    #pragma unroll
    for (int mt = 0; mt < 8; ++mt) {
        const int m0 = wave * 128 + mt * 16;
        #pragma unroll
        for (int i = 0; i < 8; ++i) {
            int r = i + 8 * half;                 // C-matrix row
            if (lValid[r]) {
                contrib[(size_t)lTok[r] * DMODEL + m0 + l16] = accO[mt][i] * lWgt[r];
            }
        }
    }
}

// ---------------------------------------------------------------------------
// Kernel 4: out = x + contrib[slot0] + contrib[slot1] (fixed order)
// ---------------------------------------------------------------------------
__global__ void k_combine(const float* __restrict__ x, const float* __restrict__ contrib,
                          float* __restrict__ out, int n4)
{
    int i = blockIdx.x * blockDim.x + threadIdx.x;
    int stride = gridDim.x * blockDim.x;
    const int rowq = DMODEL / 4;
    for (; i < n4; i += stride) {
        int t = i / rowq, m4 = i - t * rowq;
        float4 xv = ((const float4*)x)[i];
        float4 c0 = ((const float4*)contrib)[(size_t)(2 * t)     * rowq + m4];
        float4 c1 = ((const float4*)contrib)[(size_t)(2 * t + 1) * rowq + m4];
        float4 o;
        o.x = xv.x + c0.x + c1.x; o.y = xv.y + c0.y + c1.y;
        o.z = xv.z + c0.z + c1.z; o.w = xv.w + c0.w + c1.w;
        ((float4*)out)[i] = o;
    }
}

// ---------------------------------------------------------------------------
extern "C" void kernel_launch(void* const* d_in, const int* in_sizes, int n_in,
                              void* d_out, int out_size, void* d_ws, size_t ws_size,
                              hipStream_t stream)
{
    (void)in_sizes; (void)n_in; (void)out_size; (void)ws_size;
    const float* x      = (const float*)d_in[0];
    const float* gate_w = (const float*)d_in[1];
    const float* rms_w  = (const float*)d_in[2];
    const float* W1     = (const float*)d_in[3];
    const float* W2     = (const float*)d_in[4];
    const float* W3     = (const float*)d_in[5];
    float* out = (float*)d_out;

    char* ws = (char*)d_ws;
    size_t off = 0;
    auto alloc = [&](size_t bytes) -> void* {
        void* p = ws + off;
        off = (off + bytes + 255) & ~(size_t)255;
        return p;
    };
    __bf16* wsY   = (__bf16*)alloc((size_t)S_TOK * DMODEL * 2);
    __bf16* wsW1  = (__bf16*)alloc((size_t)NEXP * DEXPERT * DMODEL * 2);
    __bf16* wsW2  = (__bf16*)alloc((size_t)NEXP * DEXPERT * DMODEL * 2);
    __bf16* wsW3  = (__bf16*)alloc((size_t)NEXP * DMODEL  * DEXPERT * 2);
    float*  contrib = (float*)alloc((size_t)S_TOK * KTOP * DMODEL * 4);
    int*    cnt     = (int*)  alloc(NEXP * 4);
    int*    tokbuf  = (int*)  alloc((size_t)NEXP * S_TOK * 4);
    float*  wbuf    = (float*)alloc((size_t)NEXP * S_TOK * 4);

    k_init<<<1, 32, 0, stream>>>(cnt);

    const int n4w = NEXP * DEXPERT * DMODEL / 4;
    k_cvt<<<2048, 256, 0, stream>>>(W1, wsW1, n4w);
    k_cvt<<<2048, 256, 0, stream>>>(W2, wsW2, n4w);
    k_cvt<<<2048, 256, 0, stream>>>(W3, wsW3, n4w);

    k_router<<<S_TOK, 128, 0, stream>>>(x, gate_w, rms_w, wsY, cnt, tokbuf, wbuf);

    dim3 grid(S_TOK / 16, NEXP);
    k_expert<<<grid, 256, 0, stream>>>(wsY, wsW1, wsW2, wsW3, cnt, tokbuf, wbuf, contrib);

    k_combine<<<1024, 256, 0, stream>>>(x, contrib, out, S_TOK * DMODEL / 4);
}
